// DeepSeekMOE_88605175316749
// MI455X (gfx1250) — matrix operand order repaired and checked by reading, not compiled, not run on hardware
//
#include <hip/hip_runtime.h>
#include <hip/hip_bf16.h>

// ---------------------------------------------------------------------------
// DeepSeek MoE grouped FFN for gfx1250 (MI455X), bf16 WMMA path.
//
// Fast path (ws_size >= 223 MB):
//   P1: x (f32) -> x_bf16
//   P2: w_up_gate, w_down (f32, [k][n]) -> bf16 transposed [n][k]
//   K1: h = x @ w_up_gate, fused SwiGLU -> act bf16
//       (tiles fetched with global_load_async_to_lds_b128, ASYNCcnt sync)
//   K2: out = act @ w_down (f32 out), same async tile path
// Fallback (ws_size >= 34.6 MB): in-loop f32->bf16 conversion variants.
// ---------------------------------------------------------------------------

typedef __attribute__((ext_vector_type(8)))  float   v8f;
typedef __attribute__((ext_vector_type(2)))  __bf16  bf16x2;
typedef __attribute__((ext_vector_type(8)))  __bf16  bf16x8;
typedef __attribute__((ext_vector_type(16))) __bf16  bf16x16;

#define G_GROUPS 8
#define M_DIM    1536
#define H_DIM    2048
#define E_DIM    1408
#define TWOE_DIM 2816

__device__ __forceinline__ unsigned pk_bf16(float a, float b) {
#if __has_builtin(__builtin_amdgcn_cvt_pk_bf16_f32)
    union { bf16x2 v; unsigned u; } cv;
    cv.v = __builtin_amdgcn_cvt_pk_bf16_f32(a, b);
    return cv.u;
#else
    unsigned ua = __float_as_uint(a);
    unsigned ub = __float_as_uint(b);
    ua += 0x7FFFu + ((ua >> 16) & 1u);
    ub += 0x7FFFu + ((ub >> 16) & 1u);
    return (ua >> 16) | (ub & 0xFFFF0000u);
#endif
}

__device__ __forceinline__ unsigned short f2bf(float f) {
    return (unsigned short)(pk_bf16(f, 0.0f) & 0xFFFFu);
}

// CDNA5 async global->LDS copy (VGLOBAL GLOBAL_LOAD_ASYNC_TO_LDS_B128).
// Per lane: 16 bytes MEM[gaddr] -> LDS[lds_byte_off]. Tracked by ASYNCcnt.
__device__ __forceinline__ void async_load_b128(unsigned lds_byte_off,
                                                const void* gaddr) {
    asm volatile("global_load_async_to_lds_b128 %0, %1, off"
                 :: "v"(lds_byte_off), "v"(gaddr)
                 : "memory");
}

__device__ __forceinline__ void wait_async0() {
#if __has_builtin(__builtin_amdgcn_s_wait_asynccnt)
    __builtin_amdgcn_s_wait_asynccnt(0);
#else
    asm volatile("s_wait_asynccnt 0" ::: "memory");
#endif
}

// Low 32 bits of a generic pointer to __shared__ = LDS byte offset.
__device__ __forceinline__ unsigned lds_off(const void* p) {
    return (unsigned)(size_t)p;
}

// A fragment (16x32 bf16 MxK): lanes 0-15 -> M, half=lane/16.
// half selects contiguous K range [half*16, half*16+15].
__device__ __forceinline__ bf16x16 lds_frag_a(const unsigned short* p) {
    bf16x8 lo = *(const bf16x8*)(p);
    bf16x8 hi = *(const bf16x8*)(p + 8);
    bf16x16 r;
#pragma unroll
    for (int i = 0; i < 8; ++i) { r[i] = lo[i]; r[i + 8] = hi[i]; }
    return r;
}

// B fragment (32x16 bf16 KxN) from N-major tile: lane%16 -> N,
// half selects contiguous K range [half*16, half*16+15].
__device__ __forceinline__ bf16x16 lds_frag_b(const unsigned short* p) {
    bf16x8 lo = *(const bf16x8*)(p);
    bf16x8 hi = *(const bf16x8*)(p + 8);
    bf16x16 r;
#pragma unroll
    for (int i = 0; i < 8; ++i) { r[i] = lo[i]; r[i + 8] = hi[i]; }
    return r;
}

__device__ __forceinline__ v8f wmma_bf16(bf16x16 a, bf16x16 b, v8f c) {
    return __builtin_amdgcn_wmma_f32_16x16x32_bf16(
        false, a, false, b, (short)0, c, false, false);
}

// ===========================================================================
// Pre-pass kernels
// ===========================================================================

__global__ __launch_bounds__(256) void convert_x_kernel(
    const float* __restrict__ src, unsigned short* __restrict__ dst)
{
    const size_t idx = (size_t)blockIdx.x * 256 + threadIdx.x;
    const float4 v = ((const float4*)src)[idx];
    uint2 d;
    d.x = pk_bf16(v.x, v.y);
    d.y = pk_bf16(v.z, v.w);
    ((uint2*)dst)[idx] = d;
}

// Per-group transpose+convert: src f32 [g][K][N] -> dst bf16 [g][N][K].
__global__ __launch_bounds__(256) void transpose_w_kernel(
    const float* __restrict__ src, unsigned short* __restrict__ dst,
    int K, int N)
{
    constexpr int LDT = 36;
    __shared__ unsigned short t[32 * LDT];
    const size_t gofs = (size_t)blockIdx.z * K * N;
    const int k0 = blockIdx.y * 32;
    const int n0 = blockIdx.x * 32;
    const int tid = threadIdx.x;

    const int rk = tid >> 3, rn4 = (tid & 7) << 2;
    const float4 v =
        *(const float4*)(src + gofs + (size_t)(k0 + rk) * N + n0 + rn4);
    t[(rn4 + 0) * LDT + rk] = f2bf(v.x);
    t[(rn4 + 1) * LDT + rk] = f2bf(v.y);
    t[(rn4 + 2) * LDT + rk] = f2bf(v.z);
    t[(rn4 + 3) * LDT + rk] = f2bf(v.w);
    __syncthreads();

    const int wn = tid >> 3, wk4 = (tid & 7) << 2;
    const uint2 d = *(const uint2*)&t[wn * LDT + wk4];
    *(uint2*)(dst + gofs + (size_t)(n0 + wn) * K + k0 + wk4) = d;
}

// ===========================================================================
// Fast-path GEMMs (all-bf16 operands, async global->LDS tile fills)
// ===========================================================================

__global__ __launch_bounds__(256) void moe_upgate_swiglu_pre_kernel(
    const unsigned short* __restrict__ xb,   // [G, M, H] bf16
    const unsigned short* __restrict__ wT,   // [G, 2E, H] bf16 (N-major)
    unsigned short* __restrict__ act)        // [G, M, E] bf16
{
    constexpr int BM = 128, BK = 32, BN = 64, LD = 40;
    __shared__ unsigned short sA [2][BM * LD];
    __shared__ unsigned short sBg[2][BN * LD];
    __shared__ unsigned short sBu[2][BN * LD];

    const int g  = blockIdx.z;
    const int m0 = blockIdx.y * BM;
    const int e0 = blockIdx.x * BN;
    const int tid   = threadIdx.x;
    const int lane  = tid & 31;
    const int wid   = tid >> 5;
    const int waveM = wid >> 2;
    const int waveN = wid & 3;
    const int half  = lane >> 4;
    const int lrow  = lane & 15;

    const unsigned short* xg = xb + (size_t)g * M_DIM * H_DIM + (size_t)m0 * H_DIM;
    const unsigned short* wg = wT + (size_t)g * TWOE_DIM * H_DIM;

    const int a_m  = tid >> 2;          // + j*64
    const int a_k8 = (tid & 3) << 3;
    const unsigned short* a_ptr = xg + (size_t)a_m * H_DIM + a_k8;

    const int b_n  = tid >> 2;          // 0..63
    const int b_k8 = (tid & 3) << 3;
    const unsigned short* bg_ptr = wg + (size_t)(e0 + b_n) * H_DIM + b_k8;
    const unsigned short* bu_ptr = bg_ptr + (size_t)E_DIM * H_DIM;

    auto load_tiles = [&](int k0, int buf) {
#pragma unroll
        for (int j = 0; j < 2; ++j)
            async_load_b128(lds_off(&sA[buf][(a_m + j * 64) * LD + a_k8]),
                            a_ptr + (size_t)j * 64 * H_DIM + k0);
        async_load_b128(lds_off(&sBg[buf][b_n * LD + b_k8]), bg_ptr + k0);
        async_load_b128(lds_off(&sBu[buf][b_n * LD + b_k8]), bu_ptr + k0);
    };

    v8f accG[4], accU[4];
#pragma unroll
    for (int i = 0; i < 4; ++i) { accG[i] = {}; accU[i] = {}; }

    load_tiles(0, 0);
    wait_async0();
    __syncthreads();

    int p = 0;
    for (int k0 = 0; k0 < H_DIM; k0 += BK) {
        if (k0 + BK < H_DIM) load_tiles(k0 + BK, p ^ 1);

        const bf16x16 bg =
            lds_frag_b(&sBg[p][(waveN * 16 + lrow) * LD + half * 16]);
        const bf16x16 bu =
            lds_frag_b(&sBu[p][(waveN * 16 + lrow) * LD + half * 16]);
#pragma unroll
        for (int mf = 0; mf < 4; ++mf) {
            const bf16x16 a =
                lds_frag_a(&sA[p][(waveM * 64 + mf * 16 + lrow) * LD + half * 16]);
            accG[mf] = wmma_bf16(a, bg, accG[mf]);
            accU[mf] = wmma_bf16(a, bu, accU[mf]);
        }
        wait_async0();
        __syncthreads();
        p ^= 1;
    }

    unsigned short* actg =
        act + (size_t)g * M_DIM * E_DIM + (size_t)m0 * E_DIM + e0;
    const int nn = waveN * 16 + lrow;
#pragma unroll
    for (int mf = 0; mf < 4; ++mf) {
#pragma unroll
        for (int r = 0; r < 8; ++r) {
            const int m = waveM * 64 + mf * 16 + half * 8 + r;
            const float gv = accG[mf][r];
            const float uv = accU[mf][r];
            const float sv = __builtin_amdgcn_rcpf(1.0f + __expf(-gv));
            actg[(size_t)m * E_DIM + nn] = f2bf(gv * sv * uv);
        }
    }
}

__global__ __launch_bounds__(256) void moe_down_pre_kernel(
    const unsigned short* __restrict__ act,  // [G, M, E] bf16
    const unsigned short* __restrict__ wT,   // [G, H, E] bf16 (N-major)
    float* __restrict__ out)                 // [G, M, H]
{
    constexpr int BM = 128, BK = 32, BN = 128, LD = 40;
    __shared__ unsigned short sA[2][BM * LD];
    __shared__ unsigned short sB[2][BN * LD];

    const int g  = blockIdx.z;
    const int m0 = blockIdx.y * BM;
    const int n0 = blockIdx.x * BN;
    const int tid   = threadIdx.x;
    const int lane  = tid & 31;
    const int wid   = tid >> 5;
    const int waveM = wid >> 2;
    const int waveN = wid & 3;
    const int half  = lane >> 4;
    const int lrow  = lane & 15;

    const unsigned short* actg =
        act + (size_t)g * M_DIM * E_DIM + (size_t)m0 * E_DIM;
    const unsigned short* wg = wT + (size_t)g * H_DIM * E_DIM;

    const int a_m  = tid >> 2;          // + j*64
    const int a_k8 = (tid & 3) << 3;
    const unsigned short* a_ptr = actg + (size_t)a_m * E_DIM + a_k8;

    const int b_n  = tid >> 2;          // + j*64
    const int b_k8 = (tid & 3) << 3;
    const unsigned short* b_ptr = wg + (size_t)(n0 + b_n) * E_DIM + b_k8;

    auto load_tiles = [&](int k0, int buf) {
#pragma unroll
        for (int j = 0; j < 2; ++j) {
            async_load_b128(lds_off(&sA[buf][(a_m + j * 64) * LD + a_k8]),
                            a_ptr + (size_t)j * 64 * E_DIM + k0);
            async_load_b128(lds_off(&sB[buf][(b_n + j * 64) * LD + b_k8]),
                            b_ptr + (size_t)j * 64 * E_DIM + k0);
        }
    };

    v8f acc[4][2];
#pragma unroll
    for (int i = 0; i < 4; ++i) { acc[i][0] = {}; acc[i][1] = {}; }

    load_tiles(0, 0);
    wait_async0();
    __syncthreads();

    int p = 0;
    for (int k0 = 0; k0 < E_DIM; k0 += BK) {
        if (k0 + BK < E_DIM) load_tiles(k0 + BK, p ^ 1);

        const bf16x16 b0 =
            lds_frag_b(&sB[p][(waveN * 32 + 0  + lrow) * LD + half * 16]);
        const bf16x16 b1 =
            lds_frag_b(&sB[p][(waveN * 32 + 16 + lrow) * LD + half * 16]);
#pragma unroll
        for (int mf = 0; mf < 4; ++mf) {
            const bf16x16 a =
                lds_frag_a(&sA[p][(waveM * 64 + mf * 16 + lrow) * LD + half * 16]);
            acc[mf][0] = wmma_bf16(a, b0, acc[mf][0]);
            acc[mf][1] = wmma_bf16(a, b1, acc[mf][1]);
        }
        wait_async0();
        __syncthreads();
        p ^= 1;
    }

    float* og = out + (size_t)g * M_DIM * H_DIM + (size_t)m0 * H_DIM + n0;
#pragma unroll
    for (int mf = 0; mf < 4; ++mf) {
#pragma unroll
        for (int nf = 0; nf < 2; ++nf) {
            const int n = waveN * 32 + nf * 16 + lrow;
#pragma unroll
            for (int r = 0; r < 8; ++r) {
                const int m = waveM * 64 + mf * 16 + half * 8 + r;
                og[(size_t)m * H_DIM + n] = acc[mf][nf][r];
            }
        }
    }
}

// ===========================================================================
// Fallback GEMMs (in-loop f32->bf16 conversion; ws only holds act)
// ===========================================================================

__global__ __launch_bounds__(256) void moe_upgate_swiglu_raw_kernel(
    const float* __restrict__ x, const float* __restrict__ w,
    unsigned short* __restrict__ act)
{
    constexpr int BM = 128, BK = 32, BN = 64, LD = 40;
    __shared__ unsigned short sA [2][BM * LD];
    __shared__ unsigned short sBg[2][BN * LD];
    __shared__ unsigned short sBu[2][BN * LD];

    const int g  = blockIdx.z;
    const int m0 = blockIdx.y * BM;
    const int e0 = blockIdx.x * BN;
    const int tid   = threadIdx.x;
    const int lane  = tid & 31;
    const int wid   = tid >> 5;
    const int waveM = wid >> 2;
    const int waveN = wid & 3;
    const int half  = lane >> 4;
    const int lrow  = lane & 15;

    const float* xg = x + (size_t)g * M_DIM * H_DIM + (size_t)m0 * H_DIM;
    const float* wg = w + (size_t)g * H_DIM * TWOE_DIM;

    const int a_mb = tid >> 3;
    const int a_kk = (tid & 7) << 2;
    const float* a_ptr = xg + (size_t)a_mb * H_DIM + a_kk;

    const int b_k = (tid >> 5) << 1;
    const int b_n = (tid & 31) << 1;
    const float* b_ptr = wg + (size_t)b_k * TWOE_DIM + e0 + b_n;

    auto load_tiles = [&](int k0, int buf) {
#pragma unroll
        for (int j = 0; j < 4; ++j) {
            const float4 v = *(const float4*)(a_ptr + (size_t)j * 32 * H_DIM + k0);
            uint2 d;
            d.x = pk_bf16(v.x, v.y);
            d.y = pk_bf16(v.z, v.w);
            *(uint2*)&sA[buf][(a_mb + j * 32) * LD + a_kk] = d;
        }
#pragma unroll
        for (int j = 0; j < 2; ++j) {
            const int k = b_k + j * 16;
            const float* r0 = b_ptr + (size_t)(k0 + j * 16) * TWOE_DIM;
            const float* r1 = r0 + TWOE_DIM;
            const float2 g0 = *(const float2*)(r0);
            const float2 g1 = *(const float2*)(r1);
            const float2 u0 = *(const float2*)(r0 + E_DIM);
            const float2 u1 = *(const float2*)(r1 + E_DIM);
            *(unsigned*)&sBg[buf][(b_n + 0) * LD + k] = pk_bf16(g0.x, g1.x);
            *(unsigned*)&sBg[buf][(b_n + 1) * LD + k] = pk_bf16(g0.y, g1.y);
            *(unsigned*)&sBu[buf][(b_n + 0) * LD + k] = pk_bf16(u0.x, u1.x);
            *(unsigned*)&sBu[buf][(b_n + 1) * LD + k] = pk_bf16(u0.y, u1.y);
        }
    };

    v8f accG[4], accU[4];
#pragma unroll
    for (int i = 0; i < 4; ++i) { accG[i] = {}; accU[i] = {}; }

    load_tiles(0, 0);
    int p = 0;
    for (int k0 = 0; k0 < H_DIM; k0 += BK) {
        __syncthreads();
        if (k0 + BK < H_DIM) load_tiles(k0 + BK, p ^ 1);

        const bf16x16 bg =
            lds_frag_b(&sBg[p][(waveN * 16 + lrow) * LD + half * 16]);
        const bf16x16 bu =
            lds_frag_b(&sBu[p][(waveN * 16 + lrow) * LD + half * 16]);
#pragma unroll
        for (int mf = 0; mf < 4; ++mf) {
            const bf16x16 a =
                lds_frag_a(&sA[p][(waveM * 64 + mf * 16 + lrow) * LD + half * 16]);
            accG[mf] = wmma_bf16(a, bg, accG[mf]);
            accU[mf] = wmma_bf16(a, bu, accU[mf]);
        }
        p ^= 1;
    }

    unsigned short* actg =
        act + (size_t)g * M_DIM * E_DIM + (size_t)m0 * E_DIM + e0;
    const int nn = waveN * 16 + lrow;
#pragma unroll
    for (int mf = 0; mf < 4; ++mf) {
#pragma unroll
        for (int r = 0; r < 8; ++r) {
            const int m = waveM * 64 + mf * 16 + half * 8 + r;
            const float gv = accG[mf][r];
            const float uv = accU[mf][r];
            const float sv = __builtin_amdgcn_rcpf(1.0f + __expf(-gv));
            actg[(size_t)m * E_DIM + nn] = f2bf(gv * sv * uv);
        }
    }
}

__global__ __launch_bounds__(256) void moe_down_raw_kernel(
    const unsigned short* __restrict__ act, const float* __restrict__ w,
    float* __restrict__ out)
{
    constexpr int BM = 128, BK = 32, BN = 128, LD = 40;
    __shared__ unsigned short sA[2][BM * LD];
    __shared__ unsigned short sB[2][BN * LD];

    const int g  = blockIdx.z;
    const int m0 = blockIdx.y * BM;
    const int n0 = blockIdx.x * BN;
    const int tid   = threadIdx.x;
    const int lane  = tid & 31;
    const int wid   = tid >> 5;
    const int waveM = wid >> 2;
    const int waveN = wid & 3;
    const int half  = lane >> 4;
    const int lrow  = lane & 15;

    const unsigned short* actg =
        act + (size_t)g * M_DIM * E_DIM + (size_t)m0 * E_DIM;
    const float* wg = w + (size_t)g * E_DIM * H_DIM;

    const int a_mb = tid >> 2;
    const int a_k8 = (tid & 3) << 3;
    const unsigned short* a_ptr = actg + (size_t)a_mb * E_DIM + a_k8;

    const int b_k = (tid >> 6) << 1;
    const int b_n = (tid & 63) << 1;
    const float* b_ptr = wg + (size_t)b_k * H_DIM + n0 + b_n;

    auto load_tiles = [&](int k0, int buf) {
#pragma unroll
        for (int j = 0; j < 2; ++j) {
            const uint4 v = *(const uint4*)(a_ptr + (size_t)j * 64 * E_DIM + k0);
            *(uint4*)&sA[buf][(a_mb + j * 64) * LD + a_k8] = v;
        }
#pragma unroll
        for (int j = 0; j < 4; ++j) {
            const int k = b_k + j * 8;
            const float* r0 = b_ptr + (size_t)(k0 + j * 8) * H_DIM;
            const float* r1 = r0 + H_DIM;
            const float2 v0 = *(const float2*)(r0);
            const float2 v1 = *(const float2*)(r1);
            *(unsigned*)&sB[buf][(b_n + 0) * LD + k] = pk_bf16(v0.x, v1.x);
            *(unsigned*)&sB[buf][(b_n + 1) * LD + k] = pk_bf16(v0.y, v1.y);
        }
    };

    v8f acc[4][2];
#pragma unroll
    for (int i = 0; i < 4; ++i) { acc[i][0] = {}; acc[i][1] = {}; }

    load_tiles(0, 0);
    int p = 0;
    for (int k0 = 0; k0 < E_DIM; k0 += BK) {
        __syncthreads();
        if (k0 + BK < E_DIM) load_tiles(k0 + BK, p ^ 1);

        const bf16x16 b0 =
            lds_frag_b(&sB[p][(waveN * 32 + 0  + lrow) * LD + half * 16]);
        const bf16x16 b1 =
            lds_frag_b(&sB[p][(waveN * 32 + 16 + lrow) * LD + half * 16]);
#pragma unroll
        for (int mf = 0; mf < 4; ++mf) {
            const bf16x16 a =
                lds_frag_a(&sA[p][(waveM * 64 + mf * 16 + lrow) * LD + half * 16]);
            acc[mf][0] = wmma_bf16(a, b0, acc[mf][0]);
            acc[mf][1] = wmma_bf16(a, b1, acc[mf][1]);
        }
        p ^= 1;
    }

    float* og = out + (size_t)g * M_DIM * H_DIM + (size_t)m0 * H_DIM + n0;
#pragma unroll
    for (int mf = 0; mf < 4; ++mf) {
#pragma unroll
        for (int nf = 0; nf < 2; ++nf) {
            const int n = waveN * 32 + nf * 16 + lrow;
#pragma unroll
            for (int r = 0; r < 8; ++r) {
                const int m = waveM * 64 + mf * 16 + half * 8 + r;
                og[(size_t)m * H_DIM + n] = acc[mf][nf][r];
            }
        }
    }
}

// ---------------------------------------------------------------------------
extern "C" void kernel_launch(void* const* d_in, const int* in_sizes, int n_in,
                              void* d_out, int out_size, void* d_ws, size_t ws_size,
                              hipStream_t stream) {
    (void)in_sizes; (void)n_in; (void)out_size;
    const float* x         = (const float*)d_in[0];  // [8,1536,2048]
    const float* w_up_gate = (const float*)d_in[1];  // [8,2048,2816]
    const float* w_down    = (const float*)d_in[2];  // [8,1408,2048]
    float* out = (float*)d_out;                      // [8,1536,2048]

    const size_t X_E   = (size_t)G_GROUPS * M_DIM * H_DIM;
    const size_t WUG_E = (size_t)G_GROUPS * H_DIM * TWOE_DIM;
    const size_t WD_E  = (size_t)G_GROUPS * E_DIM * H_DIM;
    const size_t ACT_E = (size_t)G_GROUPS * M_DIM * E_DIM;
    const size_t FULL_BYTES = (X_E + WUG_E + WD_E + ACT_E) * 2;

    dim3 block(256);
    dim3 grid1(E_DIM / 64,  M_DIM / 128, G_GROUPS);   // 22 x 12 x 8
    dim3 grid2(H_DIM / 128, M_DIM / 128, G_GROUPS);   // 16 x 12 x 8

    if (ws_size >= FULL_BYTES) {
        unsigned short* xb   = (unsigned short*)d_ws;
        unsigned short* wTug = xb + X_E;
        unsigned short* wTd  = wTug + WUG_E;
        unsigned short* act  = wTd + WD_E;

        convert_x_kernel<<<dim3((unsigned)(X_E / 4 / 256)), block, 0, stream>>>(x, xb);
        transpose_w_kernel<<<dim3(TWOE_DIM / 32, H_DIM / 32, G_GROUPS),
                             block, 0, stream>>>(w_up_gate, wTug, H_DIM, TWOE_DIM);
        transpose_w_kernel<<<dim3(H_DIM / 32, E_DIM / 32, G_GROUPS),
                             block, 0, stream>>>(w_down, wTd, E_DIM, H_DIM);

        moe_upgate_swiglu_pre_kernel<<<grid1, block, 0, stream>>>(xb, wTug, act);
        moe_down_pre_kernel<<<grid2, block, 0, stream>>>(act, wTd, out);
    } else {
        unsigned short* act = (unsigned short*)d_ws;  // 34.6 MB
        moe_upgate_swiglu_raw_kernel<<<grid1, block, 0, stream>>>(x, w_up_gate, act);
        moe_down_raw_kernel<<<grid2, block, 0, stream>>>(act, w_down, out);
    }
}
